// Decoder_86921548137026
// MI455X (gfx1250) — compile-verified, matrix operand order checked by reading, not run on hardware
//
#include <hip/hip_runtime.h>
#include <hip/hip_bf16.h>

// ---------------- problem constants ----------------
#define H_    200
#define P_    8
#define B_    32
#define L_    1024
#define TWOH  400          // 2*H
#define HP    1600         // H*P
#define FOURH 800          // 4*H
#define K1P   416          // 400 padded to mult of 32
#define K2P   224          // 200 padded to mult of 32
#define ROWS  32768        // B*L
#define BIGF  1e30f

typedef __attribute__((ext_vector_type(16))) __bf16    v16bf;
typedef __attribute__((ext_vector_type(8)))  float     v8f;
typedef __attribute__((ext_vector_type(4)))  unsigned  v4u;   // 16B, 16B-aligned

// ---------------- helpers ----------------
static __device__ __forceinline__ unsigned short f2bf(float f) {
  unsigned u = __float_as_uint(f);
  u = (u + 0x7FFFu + ((u >> 16) & 1u)) >> 16;   // round-to-nearest-even
  return (unsigned short)u;
}
static __device__ __forceinline__ float bf2f(unsigned short h) {
  return __uint_as_float(((unsigned)h) << 16);
}
static __device__ __forceinline__ float sigmoidf_(float x) {
  return 1.0f / (1.0f + __expf(-x));
}

// Async global->LDS copy of 16 bytes (per lane). CDNA5: tracked by ASYNCcnt.
// LDS destination address = low 32 bits of the flat shared pointer.
static __device__ __forceinline__ void stage_copy16(unsigned short* ldsDst,
                                                    const unsigned short* gsrc) {
  unsigned lo = (unsigned)(size_t)ldsDst;
  asm volatile("global_load_async_to_lds_b128 %0, %1, off"
               :: "v"(lo), "v"(gsrc)
               : "memory");
}
#define WAIT_ASYNC(n) asm volatile("s_wait_asynccnt %0" :: "i"(n) : "memory")

// ---------------- one-time prep: transpose/convert to bf16 ----------------
// enc (B, 2H, L) f32  ->  Ebf (B*L, K1P) bf16 row-major, K-padded with zeros
__global__ void k_prep_E(const float* __restrict__ enc, unsigned short* __restrict__ Ebf) {
  int i = blockIdx.x * blockDim.x + threadIdx.x;   // exactly B_*K1P*L_ threads
  int l = i & (L_ - 1);
  int rest = i >> 10;
  int c = rest % K1P;
  int b = rest / K1P;
  if (b >= B_) return;
  float v = (c < TWOH) ? enc[((size_t)b * TWOH + c) * L_ + l] : 0.0f;
  Ebf[((size_t)(b * L_ + l)) * K1P + c] = f2bf(v);
}

// src (K, N) f32 -> dst (N, Kpad) bf16 (transposed, K-padded with zeros)
__global__ void k_conv_wt(unsigned short* __restrict__ dst, const float* __restrict__ src,
                          int K, int Kpad, int N) {
  int i = blockIdx.x * blockDim.x + threadIdx.x;
  if (i >= N * Kpad) return;
  int k = i % Kpad, n = i / Kpad;
  dst[(size_t)n * Kpad + k] = (k < K) ? f2bf(src[(size_t)k * N + n]) : (unsigned short)0;
}

__global__ void k_zero(unsigned* __restrict__ p, int n) {
  int i = blockIdx.x * blockDim.x + threadIdx.x;
  if (i < n) p[i] = 0u;
}

// hx=cx=0 ; es=ee=E[:,0,:]=enc[:,:,0]
__global__ void k_init(const float* __restrict__ enc, float* __restrict__ es, float* __restrict__ ee,
                       float* __restrict__ hx, float* __restrict__ cx) {
  int i = blockIdx.x * blockDim.x + threadIdx.x;
  if (i >= B_ * TWOH) return;
  int b = i / TWOH, c = i % TWOH;
  float v = enc[((size_t)b * TWOH + c) * L_];
  es[i] = v; ee[i] = v;
  if (i < B_ * H_) { hx[i] = 0.0f; cx[i] = 0.0f; }
}

// ---------------- small dense VALU kernels ----------------
// r = tanh([hx, es, ee] @ wd)   (B,1000)x(1000,H)
__global__ void k_r(const float* __restrict__ hx, const float* __restrict__ es,
                    const float* __restrict__ ee, const float* __restrict__ wd,
                    float* __restrict__ r) {
  int i = blockIdx.x * blockDim.x + threadIdx.x;
  if (i >= B_ * H_) return;
  int b = i / H_, h = i % H_;
  float a = 0.0f;
  for (int k = 0; k < H_; ++k)   a += hx[b * H_ + k]   * wd[(size_t)k * H_ + h];
  for (int k = 0; k < TWOH; ++k) a += es[b * TWOH + k] * wd[(size_t)(H_ + k) * H_ + h];
  for (int k = 0; k < TWOH; ++k) a += ee[b * TWOH + k] * wd[(size_t)(H_ + TWOH + k) * H_ + h];
  r[i] = tanhf(a);
}

// rb = r @ w1[2H:] + b1    (B,H)x(H,HP)
__global__ void k_rb(const float* __restrict__ r, const float* __restrict__ w1,
                     const float* __restrict__ b1, float* __restrict__ rb) {
  int i = blockIdx.x * blockDim.x + threadIdx.x;
  if (i >= B_ * HP) return;
  int b = i / HP, n = i % HP;
  float a = b1[n];
  for (int k = 0; k < H_; ++k)
    a += r[b * H_ + k] * w1[(size_t)(TWOH + k) * HP + n];
  rb[i] = a;
}

// ---------------- WMMA GEMM + maxout ----------------
// Load one 16x32 bf16 operand tile in the CDNA5 WMMA register layout.
// Per lane: two 16-byte chunks at (m*ldk + 2*kg) and (m*ldk + 32 + 2*kg) bytes,
// both provably 16B-aligned -> ds_load_b128 / global_load_b128.
static __device__ __forceinline__ v16bf load_tile(const unsigned short* base,
                                                  int ldk, int lane) {
  union { v4u q[2]; v16bf v; } t;
  int m  = lane & 15;          // row (A) / col (B)
  int kg = (lane >> 4) * 8;    // lane-half K offset (elements)
  const unsigned short* p = base + (size_t)m * ldk;
  t.q[0] = *(const v4u*)(p + kg);        // pairs k = kg+0..kg+7
  t.q[1] = *(const v4u*)(p + 16 + kg);   // pairs k = 16+kg .. 16+kg+7
  return t.v;
}

// C/D layout: vgpr r, lanes 0-15 -> (M=r, N=lane); lanes 16-31 -> (M=r+8, N=lane-16)
static __device__ __forceinline__ void maxout_store(v8f c, int lane, int row0, int col0,
                                                    const float* __restrict__ bias, int biasPerBatch,
                                                    unsigned short* __restrict__ out, int ldo) {
  int N = lane & 15;
  int Mbase = (lane >> 4) * 8;
  int col = col0 + N;
  float m[8];
#pragma unroll
  for (int r = 0; r < 8; ++r) {
    int row = row0 + Mbase + r;
    float bv = biasPerBatch ? bias[(row >> 10) * HP + col] : bias[col];
    m[r] = c[r] + bv;
  }
#pragma unroll
  for (int r = 0; r < 8; ++r) {      // max over P=8 consecutive cols (8-lane groups)
    float v = m[r];
    v = fmaxf(v, __shfl_xor(v, 1, 32));
    v = fmaxf(v, __shfl_xor(v, 2, 32));
    v = fmaxf(v, __shfl_xor(v, 4, 32));
    m[r] = v;
  }
  if ((lane & 7) == 0) {
    int h = (col0 >> 3) + ((lane >> 3) & 1);
#pragma unroll
    for (int r = 0; r < 8; ++r) {
      int row = row0 + Mbase + r;
      out[(size_t)row * ldo + h] = f2bf(m[r]);
    }
  }
}

// 4 waves per workgroup, 64x64 output tile; A/B k-slices double-buffered in LDS
// via async global->LDS copies; D = maxout_P(A @ Bt + bias) -> bf16
__global__ void __launch_bounds__(128)
gemm_maxout_kernel(const unsigned short* __restrict__ A, int lda,
                   const unsigned short* __restrict__ Bt, int ldb,
                   const float* __restrict__ bias, int biasPerBatch, int ksteps,
                   unsigned short* __restrict__ out, int ldo) {
  __shared__ __align__(16) unsigned short Asl[2][64 * 32];
  __shared__ __align__(16) unsigned short Bsl[2][64 * 32];
  int tid  = threadIdx.x;
  int lane = tid & 31;
  int wave = tid >> 5;
  int row0g = blockIdx.x * 64;
  int col0g = blockIdx.y * 64;

  // cooperative staging of one 32-wide K slice (A 64x32 + B 64x32 = 8KB)
  // 128 threads x 2 chunks x 16B per matrix; 4 async ops per thread per stage
  auto stage = [&](int ks) {
    int buf = ks & 1;
    int kb  = ks * 32;
#pragma unroll
    for (int j = 0; j < 2; ++j) {
      int c    = tid * 2 + j;     // 16-byte chunk id: 0..255
      int rrow = c >> 2;          // row within 64-row slice
      int q    = (c & 3) * 8;     // 8-element (16B) chunk within 32-k slice
      stage_copy16(&Asl[buf][rrow * 32 + q],
                   A + (size_t)(row0g + rrow) * lda + kb + q);
      stage_copy16(&Bsl[buf][rrow * 32 + q],
                   Bt + (size_t)(col0g + rrow) * ldb + kb + q);
    }
  };

  int wr0 = (wave >> 1) * 32;   // this wave's rows within the 64x64 block tile
  int wc0 = (wave & 1) * 32;    // this wave's cols

  v8f c00 = {}, c01 = {}, c10 = {}, c11 = {};
  stage(0);
  for (int ks = 0; ks < ksteps; ++ks) {
    if (ks + 1 < ksteps) {
      stage(ks + 1);      // prefetch next slice into the other buffer
      WAIT_ASYNC(4);      // wait for current slice; leave the 4 new ops in flight
    } else {
      WAIT_ASYNC(0);
    }
    __syncthreads();
    int buf = ks & 1;
    v16bf a0 = load_tile(&Asl[buf][(wr0 +  0) * 32], 32, lane);
    v16bf a1 = load_tile(&Asl[buf][(wr0 + 16) * 32], 32, lane);
    v16bf b0 = load_tile(&Bsl[buf][(wc0 +  0) * 32], 32, lane);
    v16bf b1 = load_tile(&Bsl[buf][(wc0 + 16) * 32], 32, lane);
    c00 = __builtin_amdgcn_wmma_f32_16x16x32_bf16(false, a0, false, b0, (short)0, c00, false, false);
    c01 = __builtin_amdgcn_wmma_f32_16x16x32_bf16(false, a0, false, b1, (short)0, c01, false, false);
    c10 = __builtin_amdgcn_wmma_f32_16x16x32_bf16(false, a1, false, b0, (short)0, c10, false, false);
    c11 = __builtin_amdgcn_wmma_f32_16x16x32_bf16(false, a1, false, b1, (short)0, c11, false, false);
    __syncthreads();      // all waves done reading before next stage overwrites
  }
  maxout_store(c00, lane, row0g + wr0,      col0g + wc0,      bias, biasPerBatch, out, ldo);
  maxout_store(c01, lane, row0g + wr0,      col0g + wc0 + 16, bias, biasPerBatch, out, ldo);
  maxout_store(c10, lane, row0g + wr0 + 16, col0g + wc0,      bias, biasPerBatch, out, ldo);
  maxout_store(c11, lane, row0g + wr0 + 16, col0g + wc0 + 16, bias, biasPerBatch, out, ldo);
}

// ---------------- scoring: s = max_p(m12 @ w3 + b3) - zero_one ----------------
__global__ void k_score(const unsigned short* __restrict__ m1, const unsigned short* __restrict__ m2,
                        const float* __restrict__ w3, const float* __restrict__ b3,
                        const int* __restrict__ lens, float* __restrict__ score) {
  int row = blockIdx.x * blockDim.x + threadIdx.x;
  if (row >= ROWS) return;
  int b = row >> 10, l = row & (L_ - 1);
  float s[P_];
#pragma unroll
  for (int p = 0; p < P_; ++p) s[p] = b3[p];
  const unsigned short* r1 = m1 + (size_t)row * K2P;
  const unsigned short* r2 = m2 + (size_t)row * K2P;
  for (int h = 0; h < H_; ++h) {
    float v = bf2f(r1[h]);
#pragma unroll
    for (int p = 0; p < P_; ++p) s[p] += v * w3[h * P_ + p];
  }
  for (int h = 0; h < H_; ++h) {
    float v = bf2f(r2[h]);
#pragma unroll
    for (int p = 0; p < P_; ++p) s[p] += v * w3[(H_ + h) * P_ + p];
  }
  float m = s[0];
#pragma unroll
  for (int p = 1; p < P_; ++p) m = fmaxf(m, s[p]);
  float zo = (l < lens[b]) ? 0.0f : BIGF;
  score[row] = m - zo;
}

// ---------------- per-batch argmax + column gather ----------------
__global__ void k_argmax_gather(const float* __restrict__ score, const float* __restrict__ enc,
                                float* __restrict__ dst) {
  __shared__ float sm[512];
  __shared__ int   si[512];
  int b = blockIdx.x, tid = threadIdx.x;
  float best = -3.0e38f; int bi = 0;
  for (int l = tid; l < L_; l += 512) {
    float v = score[b * L_ + l];
    if (v > best) { best = v; bi = l; }
  }
  sm[tid] = best; si[tid] = bi;
  __syncthreads();
  for (int s = 256; s > 0; s >>= 1) {
    if (tid < s) {
      if (sm[tid + s] > sm[tid] || (sm[tid + s] == sm[tid] && si[tid + s] < si[tid])) {
        sm[tid] = sm[tid + s]; si[tid] = si[tid + s];
      }
    }
    __syncthreads();
  }
  int idx = si[0];
  for (int c = tid; c < TWOH; c += 512)
    dst[b * TWOH + c] = enc[((size_t)b * TWOH + c) * L_ + idx];
}

// ---------------- LSTM cell + MLP ----------------
__global__ void k_gates(const float* __restrict__ es, const float* __restrict__ ee,
                        const float* __restrict__ hx, const float* __restrict__ w_ih,
                        const float* __restrict__ w_hh, const float* __restrict__ b_lstm,
                        float* __restrict__ cx, float* __restrict__ hxp) {
  int i = blockIdx.x * blockDim.x + threadIdx.x;
  if (i >= B_ * H_) return;
  int b = i / H_, h = i % H_;
  float g[4];
#pragma unroll
  for (int q = 0; q < 4; ++q) g[q] = b_lstm[q * H_ + h];
  for (int k = 0; k < TWOH; ++k) {
    float x = es[b * TWOH + k];
#pragma unroll
    for (int q = 0; q < 4; ++q) g[q] += x * w_ih[(size_t)k * FOURH + q * H_ + h];
  }
  for (int k = 0; k < TWOH; ++k) {
    float x = ee[b * TWOH + k];
#pragma unroll
    for (int q = 0; q < 4; ++q) g[q] += x * w_ih[(size_t)(TWOH + k) * FOURH + q * H_ + h];
  }
  for (int k = 0; k < H_; ++k) {
    float x = hx[b * H_ + k];
#pragma unroll
    for (int q = 0; q < 4; ++q) g[q] += x * w_hh[(size_t)k * FOURH + q * H_ + h];
  }
  float ig = sigmoidf_(g[0]), fg = sigmoidf_(g[1]), gg = tanhf(g[2]), og = sigmoidf_(g[3]);
  float c = fg * cx[i] + ig * gg;
  cx[i] = c;
  hxp[i] = og * tanhf(c);
}

__global__ void k_mlp(const float* __restrict__ hxp, const float* __restrict__ w_mlp,
                      const float* __restrict__ b_mlp, float* __restrict__ hx) {
  int i = blockIdx.x * blockDim.x + threadIdx.x;
  if (i >= B_ * H_) return;
  int b = i / H_, h = i % H_;
  float a = b_mlp[h];
  for (int k = 0; k < H_; ++k) a += hxp[b * H_ + k] * w_mlp[(size_t)k * H_ + h];
  hx[i] = a;
}

// ---------------- masked log-softmax over L ----------------
__global__ void k_logsoftmax(const float* __restrict__ sA, const float* __restrict__ sB,
                             const int* __restrict__ lens, float* __restrict__ out) {
  __shared__ float red[256];
  int b = blockIdx.x, arr = blockIdx.y, tid = threadIdx.x;
  const float* src = arr ? sB : sA;
  int len = lens[b];
  float x[4];
  float mx = -3.0e38f;
#pragma unroll
  for (int j = 0; j < 4; ++j) {
    int l = tid + j * 256;
    float v = (l < len) ? src[b * L_ + l] : -BIGF;
    x[j] = v;
    mx = fmaxf(mx, v);
  }
  red[tid] = mx; __syncthreads();
  for (int s = 128; s > 0; s >>= 1) { if (tid < s) red[tid] = fmaxf(red[tid], red[tid + s]); __syncthreads(); }
  mx = red[0]; __syncthreads();
  float sum = 0.0f;
#pragma unroll
  for (int j = 0; j < 4; ++j) sum += __expf(x[j] - mx);
  red[tid] = sum; __syncthreads();
  for (int s = 128; s > 0; s >>= 1) { if (tid < s) red[tid] += red[tid + s]; __syncthreads(); }
  float lse = mx + __logf(red[0]);
#pragma unroll
  for (int j = 0; j < 4; ++j)
    out[(size_t)arr * ROWS + b * L_ + tid + j * 256] = x[j] - lse;
}

// ---------------- host launcher ----------------
extern "C" void kernel_launch(void* const* d_in, const int* in_sizes, int n_in,
                              void* d_out, int out_size, void* d_ws, size_t ws_size,
                              hipStream_t stream) {
  (void)in_sizes; (void)n_in; (void)out_size; (void)ws_size;
  const float* enc    = (const float*)d_in[0];
  const int*   lens   = (const int*)  d_in[1];
  const float* wd_[2] = {(const float*)d_in[2],  (const float*)d_in[9]};
  const float* w1_[2] = {(const float*)d_in[3],  (const float*)d_in[10]};
  const float* b1_[2] = {(const float*)d_in[4],  (const float*)d_in[11]};
  const float* w2_[2] = {(const float*)d_in[5],  (const float*)d_in[12]};
  const float* b2_[2] = {(const float*)d_in[6],  (const float*)d_in[13]};
  const float* w3_[2] = {(const float*)d_in[7],  (const float*)d_in[14]};
  const float* b3_[2] = {(const float*)d_in[8],  (const float*)d_in[15]};
  const float* w_ih   = (const float*)d_in[16];
  const float* w_hh   = (const float*)d_in[17];
  const float* b_lstm = (const float*)d_in[18];
  const float* w_mlp  = (const float*)d_in[19];
  const float* b_mlp  = (const float*)d_in[20];
  float* out = (float*)d_out;

  char* ws = (char*)d_ws;
  size_t off = 0;
  auto wsalloc = [&](size_t bytes) -> void* {
    void* p = ws + off;
    off = (off + bytes + 255) & ~(size_t)255;
    return p;
  };
  unsigned short* Ebf    = (unsigned short*)wsalloc((size_t)ROWS * K1P * 2);
  unsigned short* w1t[2] = {(unsigned short*)wsalloc((size_t)HP * K1P * 2),
                            (unsigned short*)wsalloc((size_t)HP * K1P * 2)};
  unsigned short* w2t[2] = {(unsigned short*)wsalloc((size_t)HP * K2P * 2),
                            (unsigned short*)wsalloc((size_t)HP * K2P * 2)};
  unsigned short* m1     = (unsigned short*)wsalloc((size_t)ROWS * K2P * 2);
  unsigned short* m2     = (unsigned short*)wsalloc((size_t)ROWS * K2P * 2);
  float* scoreBuf[2] = {(float*)wsalloc(ROWS * 4), (float*)wsalloc(ROWS * 4)};
  float* rbuf  = (float*)wsalloc(B_ * H_ * 4);
  float* rb    = (float*)wsalloc(B_ * HP * 4);
  float* hx    = (float*)wsalloc(B_ * H_ * 4);
  float* hxp   = (float*)wsalloc(B_ * H_ * 4);
  float* cx    = (float*)wsalloc(B_ * H_ * 4);
  float* esee[2] = {(float*)wsalloc(B_ * TWOH * 4), (float*)wsalloc(B_ * TWOH * 4)};

  // one-time (per launch) prep
  k_prep_E<<<(B_ * K1P * L_) / 256, 256, 0, stream>>>(enc, Ebf);
  for (int t = 0; t < 2; ++t) {
    k_conv_wt<<<(HP * K1P + 255) / 256, 256, 0, stream>>>(w1t[t], w1_[t], TWOH, K1P, HP);
    k_conv_wt<<<(HP * K2P + 255) / 256, 256, 0, stream>>>(w2t[t], w2_[t], H_, K2P, HP);
  }
  k_zero<<<(ROWS * K2P / 2 + 255) / 256, 256, 0, stream>>>((unsigned*)m1, ROWS * K2P / 2);
  k_zero<<<(ROWS * K2P / 2 + 255) / 256, 256, 0, stream>>>((unsigned*)m2, ROWS * K2P / 2);
  k_init<<<(B_ * TWOH + 255) / 256, 256, 0, stream>>>(enc, esee[0], esee[1], hx, cx);

  for (int it = 0; it < 2; ++it) {          // NITER
    for (int t = 0; t < 2; ++t) {           // t=0: start/alphas, t=1: end/betas
      k_r <<<(B_ * H_ + 255) / 256, 256, 0, stream>>>(hx, esee[0], esee[1], wd_[t], rbuf);
      k_rb<<<(B_ * HP + 255) / 256, 256, 0, stream>>>(rbuf, w1_[t], b1_[t], rb);
      // m1 = max_P(E @ w1[:2H] + rb)
      gemm_maxout_kernel<<<dim3(ROWS / 64, HP / 64), 128, 0, stream>>>(
          Ebf, K1P, w1t[t], K1P, rb, /*biasPerBatch=*/1, /*ksteps=*/K1P / 32, m1, K2P);
      // m2 = max_P(m1 @ w2 + b2)
      gemm_maxout_kernel<<<dim3(ROWS / 64, HP / 64), 128, 0, stream>>>(
          m1, K2P, w2t[t], K2P, b2_[t], /*biasPerBatch=*/0, /*ksteps=*/K2P / 32, m2, K2P);
      k_score<<<ROWS / 256, 256, 0, stream>>>(m1, m2, w3_[t], b3_[t], lens, scoreBuf[t]);
      k_argmax_gather<<<B_, 512, 0, stream>>>(scoreBuf[t], enc, esee[t]);
    }
    k_gates<<<(B_ * H_ + 255) / 256, 256, 0, stream>>>(esee[0], esee[1], hx, w_ih, w_hh, b_lstm, cx, hxp);
    k_mlp  <<<(B_ * H_ + 255) / 256, 256, 0, stream>>>(hxp, w_mlp, b_mlp, hx);
  }
  k_logsoftmax<<<dim3(B_, 2), 256, 0, stream>>>(scoreBuf[0], scoreBuf[1], lens, out);
}
